// Qwen3MoeDecoderLayer_59485297049806
// MI455X (gfx1250) — matrix operand order repaired and checked by reading, not compiled, not run on hardware
//
#include <hip/hip_runtime.h>
#include <hip/hip_bf16.h>
#include <math.h>

#define TT 1024
#define HH 2048
#define NHEADS 32
#define NKVH 4
#define DD 128
#define EE 64
#define KTOP 8
#define FF 768
#define EPSF 1e-6f
#define THETAF 1000000.0f

typedef __bf16 bf16_t;
typedef __attribute__((ext_vector_type(16))) __bf16 v16bf;
typedef __attribute__((ext_vector_type(8)))  __bf16 v8bf;
typedef __attribute__((ext_vector_type(8)))  float  v8f;
typedef __attribute__((ext_vector_type(2)))  float  v2f;

// WMMA fragment layouts (CDNA5 ISA, wave32):
//  A 16x32 bf16 : lane m = lane%16, half = lane/16; els 0..7 -> k = el + 8*half,
//                 els 8..15 -> k = el + 8 + 8*half  (two contiguous 8-element runs)
//  B 32x16 bf16 : lane n = lane%16, half = lane/16; el -> k as for A (two contiguous 8-element runs)
//  C 16x16 f32  : element i -> row = i + 8*half, col = lane%16
//  A/B 16x16x4 f32: element v -> k = v + 2*half

__device__ __forceinline__ v8f wmma_bf16(v16bf a, v16bf b, v8f c) {
  return __builtin_amdgcn_wmma_f32_16x16x32_bf16(false, a, false, b, (short)0, c, false, false);
}
__device__ __forceinline__ v8f wmma_f32k4(v2f a, v2f b, v8f c) {
  return __builtin_amdgcn_wmma_f32_16x16x4_f32(false, a, false, b, (short)0, c, false, false);
}
__device__ __forceinline__ v8f v8f_zero() {
  v8f z;
#pragma unroll
  for (int i = 0; i < 8; ++i) z[i] = 0.f;
  return z;
}
// Build a 16-element fragment from two contiguous 8-element LDS runs.
__device__ __forceinline__ v16bf frag16(const bf16_t* p0, const bf16_t* p1) {
  v8bf lo = *(const v8bf*)p0;
  v8bf hi = *(const v8bf*)p1;
  return __builtin_shufflevector(lo, hi, 0,1,2,3,4,5,6,7,8,9,10,11,12,13,14,15);
}

// CDNA5 async global->LDS copy (128b per lane); tracked by ASYNCcnt (CDNA5 ISA).
// LDS address = addr[31:0] of the generic shared pointer (ISA flat->LDS aperture mapping).
__device__ __forceinline__ void async_copy_b128(const bf16_t* gsrc, bf16_t* ldst) {
  unsigned lds = (unsigned)(uintptr_t)ldst;
  asm volatile("global_load_async_to_lds_b128 %0, %1, off"
               :: "v"(lds), "v"(gsrc) : "memory");
}
__device__ __forceinline__ void wait_asynccnt0() {
  asm volatile("s_wait_asynccnt 0x0" ::: "memory");
}

// ---------------- RMSNorm (row per block), writes bf16 (+ optional f32) ----------------
__global__ __launch_bounds__(256) void rmsnorm_kernel(const float* __restrict__ x, const float* __restrict__ w,
                                                      bf16_t* __restrict__ outb, float* __restrict__ outf) {
  int row = blockIdx.x;
  const float* xr = x + (size_t)row * HH;
  float ss = 0.f;
#pragma unroll
  for (int i = 0; i < HH/256; ++i) { float v = xr[threadIdx.x + i*256]; ss += v*v; }
  for (int o = 16; o >= 1; o >>= 1) ss += __shfl_xor(ss, o, 32);
  __shared__ float parts[8];
  int lane = threadIdx.x & 31, wv = threadIdx.x >> 5;
  if (lane == 0) parts[wv] = ss;
  __syncthreads();
  float tot = 0.f;
#pragma unroll
  for (int i = 0; i < 8; ++i) tot += parts[i];
  float r = rsqrtf(tot / (float)HH + EPSF);
#pragma unroll
  for (int i = 0; i < HH/256; ++i) {
    int c = threadIdx.x + i*256;
    float v = xr[c] * r * w[c];
    outb[(size_t)row*HH + c] = (bf16_t)v;
    if (outf) outf[(size_t)row*HH + c] = v;
  }
}

// ---------------- f32 -> bf16 conversion (dense weights only, ~40MB total) ----------------
__global__ void f32_to_bf16_kernel(const float* __restrict__ src, bf16_t* __restrict__ dst, size_t n) {
  size_t i = (size_t)blockIdx.x * blockDim.x + threadIdx.x;
  size_t stride = (size_t)gridDim.x * blockDim.x;
  for (; i < n; i += stride) dst[i] = (bf16_t)src[i];
}

// ---------------- Generic bf16 GEMM: C(f32 MxN) = A(bf16 MxK) * B(bf16 KxN) ----------------
// 128 threads = 4 waves; 64x64 tile; wave w: rows [16w,16w+16) x 64 cols.
// A tile staged via async global->LDS; B staged transposed (contiguous fragment runs).
__global__ __launch_bounds__(128) void gemm_bf16_kernel(const bf16_t* __restrict__ A, const bf16_t* __restrict__ B,
                                                        float* __restrict__ C, int M, int N, int Kd) {
  __shared__ bf16_t As[64*32];    // [m][k]
  __shared__ bf16_t BsT[64*32];   // [n][k]
  int tid = threadIdx.x, lane = tid & 31, wv = tid >> 5;
  int half = lane >> 4, n16 = lane & 15;
  int m0 = blockIdx.y * 64, nb0 = blockIdx.x * 64;
  v8f acc[4];
#pragma unroll
  for (int nt = 0; nt < 4; ++nt) acc[nt] = v8f_zero();
  for (int kb = 0; kb < Kd; kb += 32) {
    __syncthreads();
    { // A tile: 2 async b128 per thread, no VGPR bounce
      int r = tid >> 1, kh = (tid & 1) * 16;
      const bf16_t* src = A + (size_t)(m0 + r)*Kd + kb + kh;
      async_copy_b128(src,     &As[r*32 + kh]);
      async_copy_b128(src + 8, &As[r*32 + kh + 8]);
    }
#pragma unroll
    for (int g = 0; g < 2; ++g) { // B tile: wide coalesced global read, transpose on LDS store
      int item = tid + g*128;       // 256 items = 32 k x 8 n-groups
      int k = item >> 3;
      int n0 = (item & 7) * 8;
      v8bf vb = *(const v8bf*)(B + (size_t)(kb + k)*N + nb0 + n0);
#pragma unroll
      for (int j = 0; j < 8; ++j) BsT[(n0 + j)*32 + k] = vb[j];
    }
    wait_asynccnt0();
    __syncthreads();
    const bf16_t* ap = &As[(wv*16 + n16)*32 + 8*half];
    v16bf a = frag16(ap, ap + 16);
#pragma unroll
    for (int nt = 0; nt < 4; ++nt) {
      const bf16_t* bp = &BsT[(nt*16 + n16)*32 + 8*half];
      v16bf b = frag16(bp, bp + 16);
      acc[nt] = wmma_bf16(a, b, acc[nt]);
    }
  }
#pragma unroll
  for (int nt = 0; nt < 4; ++nt)
#pragma unroll
    for (int i = 0; i < 8; ++i)
      C[(size_t)(m0 + wv*16 + i + 8*half)*N + nb0 + nt*16 + n16] = acc[nt][i];
}

// ---------------- per-(t,head) RMSNorm + RoPE, write head-major bf16 ----------------
__global__ __launch_bounds__(32) void qknorm_rope_kernel(const float* __restrict__ qkv, const float* __restrict__ nw,
                                                         const int* __restrict__ pos, bf16_t* __restrict__ out_hs,
                                                         int nheads) {
  int t = blockIdx.x, h = blockIdx.y, lane = threadIdx.x;
  const float* xr = qkv + ((size_t)t * nheads + h) * DD;
  float v0 = xr[lane], v1 = xr[lane+32], v2 = xr[lane+64], v3 = xr[lane+96];
  float ss = v0*v0 + v1*v1 + v2*v2 + v3*v3;
  for (int o = 16; o >= 1; o >>= 1) ss += __shfl_xor(ss, o, 32);
  float r = rsqrtf(ss / (float)DD + EPSF);
  v0 *= r * nw[lane]; v1 *= r * nw[lane+32]; v2 *= r * nw[lane+64]; v3 *= r * nw[lane+96];
  float p = (float)pos[t];
  float a0 = p * powf(THETAF, -(float)lane / 64.f);
  float a1 = p * powf(THETAF, -(float)(lane+32) / 64.f);
  float c0 = cosf(a0), s0 = sinf(a0), c1 = cosf(a1), s1 = sinf(a1);
  bf16_t* o_ = out_hs + ((size_t)h * TT + t) * DD;
  o_[lane]    = (bf16_t)(v0*c0 - v2*s0);
  o_[lane+64] = (bf16_t)(v2*c0 + v0*s0);
  o_[lane+32] = (bf16_t)(v1*c1 - v3*s1);
  o_[lane+96] = (bf16_t)(v3*c1 + v1*s1);
}

// ---------------- V: f32 (t,kv,d) -> bf16 head-major (kv,t,d) ----------------
__global__ void vpack_kernel(const float* __restrict__ vf, bf16_t* __restrict__ vhs) {
  int i = blockIdx.x * blockDim.x + threadIdx.x;   // T*NKV*D elements
  int d = i & (DD - 1); int th = i / DD; int h = th % NKVH; int t = th / NKVH;
  vhs[((size_t)h*TT + t)*DD + d] = (bf16_t)vf[i];
}

// ---------------- Flash attention: per (head, 64-row query block), bf16 WMMA ----------------
__global__ __launch_bounds__(128) void attn_kernel(const bf16_t* __restrict__ qhs, const bf16_t* __restrict__ khs,
                                                   const bf16_t* __restrict__ vhs, bf16_t* __restrict__ attnb,
                                                   float scale) {
  int h = blockIdx.x, qb = blockIdx.y;
  int kvh = h / (NHEADS / NKVH);
  int tid = threadIdx.x, lane = tid & 31, wv = tid >> 5;
  int half = lane >> 4, n16 = lane & 15;
  __shared__ bf16_t Qs[64*DD];      // [q][d]
  __shared__ bf16_t Ks[64*DD];      // [key][d]   (QK^T B frags contiguous along d)
  __shared__ bf16_t Vt[DD*64];      // [d][key]   (PV  B frags contiguous along key)
  __shared__ bf16_t Ps[4][16*64];   // wave-private P, [m][key]
  const bf16_t* qptr = qhs + ((size_t)h*TT + qb*64) * DD;
  for (int i = tid*8; i < 64*DD; i += 128*8) async_copy_b128(&qptr[i], &Qs[i]);
  wait_asynccnt0();
  __syncthreads();
  v16bf qfrag[4];
#pragma unroll
  for (int kc4 = 0; kc4 < 4; ++kc4) {
    const bf16_t* ap = &Qs[(wv*16 + n16)*DD + kc4*32 + 8*half];
    qfrag[kc4] = frag16(ap, ap + 16);
  }
  v8f o[8];
#pragma unroll
  for (int nt = 0; nt < 8; ++nt) o[nt] = v8f_zero();
  float mrow[8], lrow[8];
#pragma unroll
  for (int i = 0; i < 8; ++i) { mrow[i] = -1e30f; lrow[i] = 0.f; }
  for (int j = 0; j <= qb; ++j) {
    __syncthreads();
    const bf16_t* kptr = khs + ((size_t)kvh*TT + j*64) * DD;
    const bf16_t* vptr = vhs + ((size_t)kvh*TT + j*64) * DD;
    for (int i = tid*8; i < 64*DD; i += 128*8) async_copy_b128(&kptr[i], &Ks[i]);
#pragma unroll
    for (int g = 0; g < 8; ++g) {   // V: wide global read, transpose on LDS store
      int item = tid + g*128;       // 1024 items = 64 keys x 16 d-groups
      int key = item >> 4;
      int d0 = (item & 15) * 8;
      v8bf vv = *(const v8bf*)&vptr[key*DD + d0];
#pragma unroll
      for (int jj = 0; jj < 8; ++jj) Vt[(d0 + jj)*64 + key] = vv[jj];
    }
    wait_asynccnt0();
    __syncthreads();
    v8f s[4];
#pragma unroll
    for (int nt = 0; nt < 4; ++nt) s[nt] = v8f_zero();
#pragma unroll
    for (int kc4 = 0; kc4 < 4; ++kc4) {   // S = Q * K^T
#pragma unroll
      for (int nt = 0; nt < 4; ++nt) {
        const bf16_t* bp = &Ks[(nt*16 + n16)*DD + kc4*32 + 8*half];
        v16bf b = frag16(bp, bp + 16);
        s[nt] = wmma_bf16(qfrag[kc4], b, s[nt]);
      }
    }
    float mcur[8];
#pragma unroll
    for (int i = 0; i < 8; ++i) mcur[i] = -1e30f;
#pragma unroll
    for (int nt = 0; nt < 4; ++nt)
#pragma unroll
      for (int i = 0; i < 8; ++i) {
        float v = s[nt][i] * scale;
        if (j == qb) {
          int cg = j*64 + nt*16 + n16;
          int rg = qb*64 + wv*16 + i + 8*half;
          if (cg > rg) v = -1e30f;
        }
        s[nt][i] = v;
        mcur[i] = fmaxf(mcur[i], v);
      }
#pragma unroll
    for (int o2 = 1; o2 < 16; o2 <<= 1)
#pragma unroll
      for (int i = 0; i < 8; ++i) mcur[i] = fmaxf(mcur[i], __shfl_xor(mcur[i], o2, 32));
    float corr[8], rsum[8];
#pragma unroll
    for (int i = 0; i < 8; ++i) {
      float mn = fmaxf(mrow[i], mcur[i]);
      corr[i] = __expf(mrow[i] - mn);
      mrow[i] = mn;
      rsum[i] = 0.f;
    }
#pragma unroll
    for (int nt = 0; nt < 4; ++nt)
#pragma unroll
      for (int i = 0; i < 8; ++i) {
        float p = __expf(s[nt][i] - mrow[i]);
        rsum[i] += p;
        Ps[wv][(i + 8*half)*64 + nt*16 + n16] = (bf16_t)p;
      }
#pragma unroll
    for (int o2 = 1; o2 < 16; o2 <<= 1)
#pragma unroll
      for (int i = 0; i < 8; ++i) rsum[i] += __shfl_xor(rsum[i], o2, 32);
#pragma unroll
    for (int i = 0; i < 8; ++i) lrow[i] = lrow[i]*corr[i] + rsum[i];
#pragma unroll
    for (int nt = 0; nt < 8; ++nt)
#pragma unroll
      for (int i = 0; i < 8; ++i) o[nt][i] *= corr[i];
#pragma unroll
    for (int kc2 = 0; kc2 < 2; ++kc2) {   // O += P * V
      const bf16_t* ap = &Ps[wv][n16*64 + kc2*32 + 8*half];
      v16bf a = frag16(ap, ap + 16);
#pragma unroll
      for (int nt = 0; nt < 8; ++nt) {
        const bf16_t* bp = &Vt[(nt*16 + n16)*64 + kc2*32 + 8*half];
        v16bf b = frag16(bp, bp + 16);
        o[nt] = wmma_bf16(a, b, o[nt]);
      }
    }
  }
#pragma unroll
  for (int nt = 0; nt < 8; ++nt)
#pragma unroll
    for (int i = 0; i < 8; ++i) {
      int rg = qb*64 + wv*16 + i + 8*half;
      attnb[(size_t)rg * (NHEADS*DD) + h*DD + nt*16 + n16] = (bf16_t)(o[nt][i] / lrow[i]);
    }
}

// ---------------- residual add ----------------
__global__ void add_residual_kernel(const float* __restrict__ hf, const float* __restrict__ hidden,
                                    float* __restrict__ res) {
  size_t i = (size_t)blockIdx.x * blockDim.x + threadIdx.x;
  res[i] = hf[i] + hidden[i];
}

// ---------------- router: softmax + top-8 + bucketing ----------------
__global__ void zero_cnt_kernel(int* cnt) { cnt[threadIdx.x] = 0; }

__global__ __launch_bounds__(256) void router_kernel(const float* __restrict__ logits, int* cnt,
                                                     int* btok, int* bk, float* bw) {
  int t = blockIdx.x * 256 + threadIdx.x;
  if (t >= TT) return;
  float l[EE];
  float mx = -1e30f;
  for (int e = 0; e < EE; ++e) { l[e] = logits[(size_t)t*EE + e]; mx = fmaxf(mx, l[e]); }
  float sum = 0.f;
  for (int e = 0; e < EE; ++e) { l[e] = __expf(l[e] - mx); sum += l[e]; }
  int isel[KTOP]; float wsel[KTOP]; float wsum = 0.f;
  for (int k = 0; k < KTOP; ++k) {
    int best = 0; float bv = -1.f;
    for (int e = 0; e < EE; ++e) if (l[e] > bv) { bv = l[e]; best = e; }
    isel[k] = best; wsel[k] = bv / sum; wsum += wsel[k];
    l[best] = -2.f;
  }
  for (int k = 0; k < KTOP; ++k) {
    int e = isel[k];
    int p = atomicAdd(&cnt[e], 1);
    btok[e*TT + p] = t; bk[e*TT + p] = k; bw[e*TT + p] = wsel[k] / wsum;
  }
}

__global__ void prefix_kernel(const int* cnt, int* off) {
  if (threadIdx.x == 0) { int s = 0; for (int e = 0; e < EE; ++e) { off[e] = s; s += cnt[e]; } }
}

__global__ __launch_bounds__(256) void map_kernel(const int* cnt, const int* off, const int* btok, const int* bk,
                                                  const float* bw, int* tokenslot, int* slottoken, float* slotweight) {
  int e = blockIdx.x;
  int c = cnt[e], base = off[e];
  for (int p = threadIdx.x; p < c; p += 256) {
    int slot = base + p;
    int t = btok[e*TT + p];
    tokenslot[t*KTOP + bk[e*TT + p]] = slot;
    slottoken[slot] = t;
    slotweight[slot] = bw[e*TT + p];
  }
}

// ---------------- MoE stage A: act = silu(x2*Wg[e]) * (x2*Wu[e]) * w  (fp32 WMMA 16x16x4) ----------------
__global__ __launch_bounds__(128) void moe_gu_kernel(const float* __restrict__ x2f, const float* __restrict__ Wg,
                                                     const float* __restrict__ Wu, const int* cnt, const int* off,
                                                     const int* slottoken, const float* slotweight,
                                                     float* __restrict__ actbuf) {
  int e = blockIdx.x;
  int count = cnt[e];
  if (count == 0) return;
  int base = off[e];
  int tid = threadIdx.x, lane = tid & 31;
  int wv = tid >> 5, half = lane >> 4, n16 = lane & 15;
  int f = blockIdx.y*64 + wv*16 + n16;
  const float* wg = Wg + (size_t)e * HH * FF;
  const float* wu = Wu + (size_t)e * HH * FF;
  for (int chunk = 0; chunk < count; chunk += 16) {
    int rm = chunk + n16;
    int tok = (rm < count) ? slottoken[base + rm] : slottoken[base];
    const float* xr = x2f + (size_t)tok * HH;
    v8f ag = v8f_zero(), au = v8f_zero();
    for (int kb = 0; kb < HH; kb += 4) {
      int k0 = kb + 2*half;
      __builtin_prefetch(wg + (size_t)(k0 + 64)*FF + f, 0, 1);
      __builtin_prefetch(wu + (size_t)(k0 + 64)*FF + f, 0, 1);
      v2f a;  a.x  = xr[k0];                  a.y  = xr[k0+1];
      v2f bg; bg.x = wg[(size_t)k0*FF + f];   bg.y = wg[(size_t)(k0+1)*FF + f];
      v2f bu; bu.x = wu[(size_t)k0*FF + f];   bu.y = wu[(size_t)(k0+1)*FF + f];
      ag = wmma_f32k4(a, bg, ag);
      au = wmma_f32k4(a, bu, au);
    }
#pragma unroll
    for (int i = 0; i < 8; ++i) {
      int r = chunk + i + 8*half;
      if (r < count) {
        float w = slotweight[base + r];
        float g = ag[i], u = au[i];
        actbuf[(size_t)(base + r)*FF + f] = (g / (1.f + __expf(-g))) * u * w;
      }
    }
  }
}

// ---------------- MoE stage B: ypart[slot] = act[slot] * Wd[e]  (fp32 WMMA 16x16x4) ----------------
__global__ __launch_bounds__(128) void moe_down_kernel(const float* __restrict__ actbuf, const float* __restrict__ Wd,
                                                       const int* cnt, const int* off, float* __restrict__ ypart) {
  int e = blockIdx.x;
  int count = cnt[e];
  if (count == 0) return;
  int base = off[e];
  int tid = threadIdx.x, lane = tid & 31;
  int wv = tid >> 5, half = lane >> 4, n16 = lane & 15;
  int hcol = blockIdx.y*64 + wv*16 + n16;
  const float* wd = Wd + (size_t)e * FF * HH;
  for (int chunk = 0; chunk < count; chunk += 16) {
    int rm = chunk + n16;
    const float* ar = actbuf + (size_t)(base + ((rm < count) ? rm : 0))*FF;
    v8f acc = v8f_zero();
    for (int kb = 0; kb < FF; kb += 4) {
      int k0 = kb + 2*half;
      __builtin_prefetch(wd + (size_t)(k0 + 64)*HH + hcol, 0, 1);
      v2f a; a.x = ar[k0];                   a.y = ar[k0+1];
      v2f b; b.x = wd[(size_t)k0*HH + hcol]; b.y = wd[(size_t)(k0+1)*HH + hcol];
      acc = wmma_f32k4(a, b, acc);
    }
#pragma unroll
    for (int i = 0; i < 8; ++i) {
      int r = chunk + i + 8*half;
      if (r < count) ypart[(size_t)(base + r)*HH + hcol] = acc[i];
    }
  }
}

// ---------------- deterministic final combine: y[t] = sum_k ypart[slot(t,k)] ----------------
__global__ void combine_kernel(const float* __restrict__ ypart, const int* __restrict__ tokenslot,
                               float* __restrict__ y) {
  size_t i = (size_t)blockIdx.x * blockDim.x + threadIdx.x;
  int t = (int)(i / HH); int hc = (int)(i % HH);
  float s = 0.f;
#pragma unroll
  for (int k = 0; k < KTOP; ++k) s += ypart[(size_t)tokenslot[t*KTOP + k]*HH + hc];
  y[i] = s;
}

extern "C" void kernel_launch(void* const* d_in, const int* in_sizes, int n_in,
                              void* d_out, int out_size, void* d_ws, size_t ws_size,
                              hipStream_t stream) {
  const int*   positions = (const int*)  d_in[0];
  const float* hidden    = (const float*)d_in[1];
  const float* ln1_w     = (const float*)d_in[2];
  const float* ln2_w     = (const float*)d_in[3];
  const float* Wq        = (const float*)d_in[4];
  const float* Wk        = (const float*)d_in[5];
  const float* Wv        = (const float*)d_in[6];
  const float* Wo        = (const float*)d_in[7];
  const float* qn_w      = (const float*)d_in[8];
  const float* kn_w      = (const float*)d_in[9];
  const float* gate_w    = (const float*)d_in[10];
  const float* Wg        = (const float*)d_in[11];
  const float* Wu        = (const float*)d_in[12];
  const float* Wd        = (const float*)d_in[13];

  float* y   = (float*)d_out;
  float* res = y + (size_t)TT*HH;

  char* wsb = (char*)d_ws;
  size_t wo = 0;
  auto take = [&](size_t bytes) -> void* {
    void* p = wsb + wo;
    wo = (wo + bytes + 255) & ~(size_t)255;
    return p;
  };
  bf16_t* x1b   = (bf16_t*)take((size_t)TT*HH*2);
  bf16_t* Wqb   = (bf16_t*)take((size_t)HH*NHEADS*DD*2);
  bf16_t* Wkb   = (bf16_t*)take((size_t)HH*NKVH*DD*2);
  bf16_t* Wvb   = (bf16_t*)take((size_t)HH*NKVH*DD*2);
  bf16_t* Wob   = (bf16_t*)take((size_t)NHEADS*DD*HH*2);
  bf16_t* gwb   = (bf16_t*)take((size_t)HH*EE*2);
  float*  qf    = (float*) take((size_t)TT*NHEADS*DD*4);
  float*  kf    = (float*) take((size_t)TT*NKVH*DD*4);
  float*  vf    = (float*) take((size_t)TT*NKVH*DD*4);
  bf16_t* qhs   = (bf16_t*)take((size_t)TT*NHEADS*DD*2);
  bf16_t* khs   = (bf16_t*)take((size_t)TT*NKVH*DD*2);
  bf16_t* vhs   = (bf16_t*)take((size_t)TT*NKVH*DD*2);
  bf16_t* attnb = (bf16_t*)take((size_t)TT*NHEADS*DD*2);
  float*  hf    = (float*) take((size_t)TT*HH*4);
  float*  x2f   = (float*) take((size_t)TT*HH*4);
  bf16_t* x2b   = (bf16_t*)take((size_t)TT*HH*2);
  float*  logit = (float*) take((size_t)TT*EE*4);
  int*    cnt   = (int*)   take((size_t)EE*4);
  int*    offb  = (int*)   take((size_t)EE*4);
  int*    btok  = (int*)   take((size_t)EE*TT*4);
  int*    bkk   = (int*)   take((size_t)EE*TT*4);
  float*  bww   = (float*) take((size_t)EE*TT*4);
  int*    tslot = (int*)   take((size_t)TT*KTOP*4);
  int*    stok  = (int*)   take((size_t)TT*KTOP*4);
  float*  swt   = (float*) take((size_t)TT*KTOP*4);
  float*  actb  = (float*) take((size_t)TT*KTOP*FF*4);
  float*  ypart = (float*) take((size_t)TT*KTOP*HH*4);

  // 1. x1 = rmsnorm(hidden, ln1) -> bf16
  rmsnorm_kernel<<<TT, 256, 0, stream>>>(hidden, ln1_w, x1b, nullptr);

  // 2. dense weights -> bf16 (one-time ~40MB; MoE weights stay fp32)
  f32_to_bf16_kernel<<<2048, 256, 0, stream>>>(Wq, Wqb, (size_t)HH*NHEADS*DD);
  f32_to_bf16_kernel<<<512,  256, 0, stream>>>(Wk, Wkb, (size_t)HH*NKVH*DD);
  f32_to_bf16_kernel<<<512,  256, 0, stream>>>(Wv, Wvb, (size_t)HH*NKVH*DD);
  f32_to_bf16_kernel<<<2048, 256, 0, stream>>>(Wo, Wob, (size_t)NHEADS*DD*HH);
  f32_to_bf16_kernel<<<128,  256, 0, stream>>>(gate_w, gwb, (size_t)HH*EE);

  // 3. QKV projections (bf16 WMMA)
  gemm_bf16_kernel<<<dim3((NHEADS*DD)/64, TT/64), 128, 0, stream>>>(x1b, Wqb, qf, TT, NHEADS*DD, HH);
  gemm_bf16_kernel<<<dim3((NKVH*DD)/64,  TT/64), 128, 0, stream>>>(x1b, Wkb, kf, TT, NKVH*DD, HH);
  gemm_bf16_kernel<<<dim3((NKVH*DD)/64,  TT/64), 128, 0, stream>>>(x1b, Wvb, vf, TT, NKVH*DD, HH);

  // 4. per-head RMSNorm + RoPE -> head-major bf16
  qknorm_rope_kernel<<<dim3(TT, NHEADS), 32, 0, stream>>>(qf, qn_w, positions, qhs, NHEADS);
  qknorm_rope_kernel<<<dim3(TT, NKVH),   32, 0, stream>>>(kf, kn_w, positions, khs, NKVH);
  vpack_kernel<<<(TT*NKVH*DD)/256, 256, 0, stream>>>(vf, vhs);

  // 5. causal flash attention (bf16 WMMA, async K/Q staging)
  attn_kernel<<<dim3(NHEADS, TT/64), 128, 0, stream>>>(qhs, khs, vhs, attnb, 0.08838834764831845f);

  // 6. output projection + residual
  gemm_bf16_kernel<<<dim3(HH/64, TT/64), 128, 0, stream>>>(attnb, Wob, hf, TT, HH, NHEADS*DD);
  add_residual_kernel<<<(TT*HH)/256, 256, 0, stream>>>(hf, hidden, res);

  // 7. x2 = rmsnorm(residual, ln2) -> bf16 + f32
  rmsnorm_kernel<<<TT, 256, 0, stream>>>(res, ln2_w, x2b, x2f);

  // 8. router logits + top-8 + expert bucketing (deterministic slot scheme)
  gemm_bf16_kernel<<<dim3(EE/64, TT/64), 128, 0, stream>>>(x2b, gwb, logit, TT, EE, HH);
  zero_cnt_kernel<<<1, EE, 0, stream>>>(cnt);
  router_kernel<<<(TT + 255)/256, 256, 0, stream>>>(logit, cnt, btok, bkk, bww);
  prefix_kernel<<<1, 1, 0, stream>>>(cnt, offb);
  map_kernel<<<EE, 256, 0, stream>>>(cnt, offb, btok, bkk, bww, tslot, stok, swt);

  // 9. grouped MoE (fp32 WMMA, weights streamed once per expert)
  moe_gu_kernel<<<dim3(EE, FF/64), 128, 0, stream>>>(x2f, Wg, Wu, cnt, offb, stok, swt, actb);
  moe_down_kernel<<<dim3(EE, HH/64), 128, 0, stream>>>(actb, Wd, cnt, offb, ypart);
  combine_kernel<<<(TT*HH)/256, 256, 0, stream>>>(ypart, tslot, y);
}
